// CustomABlock_73366631350295
// MI455X (gfx1250) — compile-verified, matrix operand order checked by reading, not run on hardware
//
#include <hip/hip_runtime.h>

typedef __attribute__((ext_vector_type(16))) _Float16 v16h;
typedef __attribute__((ext_vector_type(8)))  _Float16 v8h;
typedef __attribute__((ext_vector_type(8)))  float    v8f;

__device__ __forceinline__ v16h cat8(v8h a, v8h b) {
  return __builtin_shufflevector(a, b, 0,1,2,3,4,5,6,7,8,9,10,11,12,13,14,15);
}

// CDNA5 16-bit A/B operand layout: lane = half*16 + idx (idx = row for A, col for B)
// v16h elems 0..7  <- K = half*8  + (0..7) ; elems 8..15 <- K = 16 + half*8 + (0..7)
// rowp points at a K-contiguous row; emits two global/ds b128 loads.
__device__ __forceinline__ v16h frag16(const _Float16* rowp, int half) {
  v8h a = *(const v8h*)(rowp + half * 8);
  v8h b = *(const v8h*)(rowp + 16 + half * 8);
  return cat8(a, b);
}

// Split-fp32 GEMM micro-op: acc += Ah*Bh + Ah*Bl + Al*Bh  (f32 accumulate)
__device__ __forceinline__ v8f wmma3(v8f acc, v16h ah, v16h al, v16h bh, v16h bl) {
  acc = __builtin_amdgcn_wmma_f32_16x16x32_f16(false, ah, false, bh, (short)0, acc, false, false);
  acc = __builtin_amdgcn_wmma_f32_16x16x32_f16(false, ah, false, bl, (short)0, acc, false, false);
  acc = __builtin_amdgcn_wmma_f32_16x16x32_f16(false, al, false, bh, (short)0, acc, false, false);
  return acc;
}

__device__ __forceinline__ float gelu_tanh(float x) {
  float x3 = x * x * x;
  return 0.5f * x * (1.0f + tanhf(0.7978845608028654f * (x + 0.044715f * x3)));
}

// low 32 bits of a flat shared pointer == LDS byte address (aperture truncation rule)
__device__ __forceinline__ unsigned lds_addr_of(const void* p) {
  return (unsigned)(unsigned long long)p;
}

// ---------------- split (fp32 -> f16 hi/lo pair) kernels ----------------

__global__ void split_kernel(const float* __restrict__ s, _Float16* __restrict__ h,
                             _Float16* __restrict__ l, size_t n) {
  size_t i = (size_t)blockIdx.x * 256 + threadIdx.x;
  if (i < n) {
    float v = s[i];
    _Float16 hv = (_Float16)v;
    h[i] = hv;
    l[i] = (_Float16)(v - (float)hv);
  }
}

__global__ void split_pad_rows(const float* __restrict__ s, _Float16* __restrict__ h,
                               _Float16* __restrict__ l, int rows_valid, int cols, int rows_pad) {
  int i = blockIdx.x * 256 + threadIdx.x;
  if (i >= rows_pad * cols) return;
  int r = i / cols, c = i % cols;
  float v = (r < rows_valid) ? s[r * cols + c] : 0.0f;
  _Float16 hv = (_Float16)v;
  h[i] = hv;
  l[i] = (_Float16)(v - (float)hv);
}

__global__ void split_pad_cols(const float* __restrict__ s, _Float16* __restrict__ h,
                               _Float16* __restrict__ l, int rows, int cols_valid, int cols_pad) {
  int i = blockIdx.x * 256 + threadIdx.x;
  if (i >= rows * cols_pad) return;
  int r = i / cols_pad, c = i % cols_pad;
  float v = (c < cols_valid) ? s[r * cols_valid + c] : 0.0f;
  _Float16 hv = (_Float16)v;
  h[i] = hv;
  l[i] = (_Float16)(v - (float)hv);
}

// ---------------- tiled transpose + split:  [b][256][4096] f32 -> [b][4096][256] f16 hi/lo ----
__global__ void __launch_bounds__(256) transpose_split(const float* __restrict__ src,
                                                       size_t srcBatchStride,
                                                       _Float16* __restrict__ hT,
                                                       _Float16* __restrict__ lT) {
  int n0 = blockIdx.x * 64;
  int c0 = blockIdx.y * 64;
  int b = blockIdx.z;
  __shared__ float tile[64][65];
  const float* sp = src + (size_t)b * srcBatchStride;
  int t = threadIdx.x;
  {  // load 64c x 64n, coalesced along n
    int c = t >> 2, nn = (t & 3) * 16;
#pragma unroll
    for (int i = 0; i < 4; ++i) {
      float4 v = *(const float4*)(sp + (size_t)(c0 + c) * 4096 + n0 + nn + i * 4);
      tile[c][nn + i * 4 + 0] = v.x;
      tile[c][nn + i * 4 + 1] = v.y;
      tile[c][nn + i * 4 + 2] = v.z;
      tile[c][nn + i * 4 + 3] = v.w;
    }
  }
  __syncthreads();
  {  // store transposed, coalesced along c, 16B vector stores
    int n = t >> 2, cc = (t & 3) * 16;
#pragma unroll
    for (int seg = 0; seg < 2; ++seg) {
      v8h hv, lv;
#pragma unroll
      for (int i = 0; i < 8; ++i) {
        float v = tile[cc + seg * 8 + i][n];
        _Float16 h = (_Float16)v;
        hv[i] = h;
        lv[i] = (_Float16)(v - (float)h);
      }
      size_t o = ((size_t)b * 4096 + n0 + n) * 256 + c0 + cc + seg * 8;
      *(v8h*)(hT + o) = hv;
      *(v8h*)(lT + o) = lv;
    }
  }
}

// ---------------- LDS-free split-fp32 WMMA GEMM (all operands K-contiguous) ----------------
// Y[b,m,n] = sum_k A[m,k]*B[b,n,k]   A:[M][K] row-major, B: transposed [b][4096][K]
// Block 64Mx64N, 8 waves (4x2), wave tile 16x32.
template <bool BIAS, bool GELU, bool RES, bool WF32, bool WSPLIT, int KDIM>
__global__ void __launch_bounds__(256) gemm_dg(
    const _Float16* __restrict__ Ah, const _Float16* __restrict__ Al,
    const _Float16* __restrict__ BhT, const _Float16* __restrict__ BlT,
    const float* __restrict__ bias, const float* __restrict__ res,   // res: [b][M][4096]
    float* __restrict__ outF,                                        // [b][M][4096]
    _Float16* __restrict__ outHT, _Float16* __restrict__ outLT,      // [b][4096][Mpad]
    int M, int Mvalid, int Mpad) {
  const int n0 = blockIdx.x * 64;
  const int m0 = blockIdx.y * 64;
  const int b = blockIdx.z;
  const int t = threadIdx.x, wave = t >> 5, lane = t & 31;
  const int half = lane >> 4, idx = lane & 15;
  const int mw = wave >> 1, nw = wave & 1;

  const _Float16* ap_h = Ah + (size_t)(m0 + mw * 16 + idx) * KDIM;
  const _Float16* ap_l = Al + (size_t)(m0 + mw * 16 + idx) * KDIM;
  const size_t brow = (size_t)b * 4096;
  const _Float16* bp0_h = BhT + (brow + n0 + nw * 32 + idx) * KDIM;
  const _Float16* bp0_l = BlT + (brow + n0 + nw * 32 + idx) * KDIM;
  const _Float16* bp1_h = bp0_h + (size_t)16 * KDIM;
  const _Float16* bp1_l = bp0_l + (size_t)16 * KDIM;

  v8f acc0 = {}, acc1 = {};
  for (int k0 = 0; k0 < KDIM; k0 += 32) {
    v16h a_h = frag16(ap_h + k0, half);
    v16h a_l = frag16(ap_l + k0, half);
    v16h b0h = frag16(bp0_h + k0, half);
    v16h b0l = frag16(bp0_l + k0, half);
    v16h b1h = frag16(bp1_h + k0, half);
    v16h b1l = frag16(bp1_l + k0, half);
    acc0 = wmma3(acc0, a_h, a_l, b0h, b0l);
    acc1 = wmma3(acc1, a_h, a_l, b1h, b1l);
  }

#pragma unroll
  for (int s = 0; s < 2; ++s) {
    v8f acc = s ? acc1 : acc0;
    int n = n0 + nw * 32 + s * 16 + idx;
    int mb = m0 + mw * 16 + half * 8;  // documented C/D layout: rows M-contiguous over r
    float vals[8];
#pragma unroll
    for (int r = 0; r < 8; ++r) {
      int m = mb + r;
      float v = acc[r];
      if (BIAS) v += (m < Mvalid) ? bias[m] : 0.0f;
      if (GELU) v = gelu_tanh(v);
      if (RES) v += res[((size_t)b * M + m) * 4096 + n];
      if (WF32) outF[((size_t)b * M + m) * 4096 + n] = v;
      vals[r] = v;
    }
    if (WSPLIT) {
      v8h hv, lv;
#pragma unroll
      for (int r = 0; r < 8; ++r) {
        _Float16 h = (_Float16)vals[r];
        hv[r] = h;
        lv[r] = (_Float16)(vals[r] - (float)h);
      }
      size_t o = (brow + n) * Mpad + mb;
      *(v8h*)(outHT + o) = hv;
      *(v8h*)(outLT + o) = lv;
    }
  }
}

// ---------------- depthwise 3x3 via async global->LDS plane staging ----------------
__global__ void __launch_bounds__(256) dwconv_kernel(const float* __restrict__ in,
                                                     const float* __restrict__ wdw,
                                                     float* __restrict__ out) {
  int plane = blockIdx.x;  // b*768 + c
  int c = plane % 768;
  __shared__ __align__(16) float sp[4096];
  const float* ip = in + (size_t)plane * 4096;
  int t = threadIdx.x;
#pragma unroll
  for (int i = 0; i < 4; ++i) {  // 256 thr x 4 x 16B = whole 16KB plane
    int e = i * 1024 + t * 4;
    unsigned lds = lds_addr_of(&sp[e]);
    unsigned long long ga = (unsigned long long)(ip + e);
    asm volatile("global_load_async_to_lds_b128 %0, %1, off" ::"v"(lds), "v"(ga) : "memory");
  }
  asm volatile("s_wait_asynccnt 0" ::: "memory");
  __syncthreads();
  const float* w = wdw + c * 9;
  float w00 = w[0], w01 = w[1], w02 = w[2];
  float w10 = w[3], w11 = w[4], w12 = w[5];
  float w20 = w[6], w21 = w[7], w22 = w[8];
#pragma unroll
  for (int j = 0; j < 16; ++j) {
    int p = j * 256 + t;
    int y = p >> 6, x = p & 63;
    float s = 0.0f;
#pragma unroll
    for (int dy = -1; dy <= 1; ++dy) {
      int yy = y + dy;
      if (yy < 0 || yy > 63) continue;
      float wr0 = (dy == -1) ? w00 : (dy == 0 ? w10 : w20);
      float wr1 = (dy == -1) ? w01 : (dy == 0 ? w11 : w21);
      float wr2 = (dy == -1) ? w02 : (dy == 0 ? w12 : w22);
      const float* row = sp + yy * 64;
      if (x > 0) s += wr0 * row[x - 1];
      s += wr1 * row[x];
      if (x < 63) s += wr2 * row[x + 1];
    }
    out[(size_t)plane * 4096 + p] = s;
  }
}

// ---------------- L2 norm over N + split (q,k keep [c][n] layout) ----------------
__global__ void __launch_bounds__(256) norm_split_kernel(const float* __restrict__ qkv,
                                                         _Float16* __restrict__ dh,
                                                         _Float16* __restrict__ dl, int which) {
  int row = blockIdx.x;  // b*256 + c
  int b = row >> 8, c = row & 255;
  const float* src = qkv + ((size_t)b * 768 + which * 256 + c) * 4096;
  __shared__ float red[256];
  int t = threadIdx.x;
  float ss = 0.0f;
  for (int i = t; i < 4096; i += 256) {
    float v = src[i];
    ss += v * v;
  }
  red[t] = ss;
  __syncthreads();
  for (int s = 128; s > 0; s >>= 1) {
    if (t < s) red[t] += red[t + s];
    __syncthreads();
  }
  float rn = 1.0f / fmaxf(sqrtf(red[0]), 1e-12f);
  _Float16* oh = dh + (size_t)row * 4096;
  _Float16* ol = dl + (size_t)row * 4096;
  for (int i = t; i < 4096; i += 256) {
    float v = src[i] * rn;
    _Float16 hv = (_Float16)v;
    oh[i] = hv;
    ol[i] = (_Float16)(v - (float)hv);
  }
}

// ---------------- channel attention: S = q k^T (K=4096), softmax ----------------
__global__ void __launch_bounds__(128) attn_kernel(
    const _Float16* __restrict__ qh, const _Float16* __restrict__ ql,
    const _Float16* __restrict__ kh, const _Float16* __restrict__ kl,
    const float* __restrict__ temperature, _Float16* __restrict__ ath,
    _Float16* __restrict__ atl) {
  int bh = blockIdx.x;
  int h = bh & 7;
  const size_t base = (size_t)bh * 32 * 4096;
  int t = threadIdx.x, wave = t >> 5, lane = t & 31, half = lane >> 4, idx = lane & 15;
  int mt = wave >> 1, nt = wave & 1;
  const _Float16* qhp = qh + base + (size_t)(mt * 16 + idx) * 4096;
  const _Float16* qlp = ql + base + (size_t)(mt * 16 + idx) * 4096;
  const _Float16* khp = kh + base + (size_t)(nt * 16 + idx) * 4096;
  const _Float16* klp = kl + base + (size_t)(nt * 16 + idx) * 4096;
  v8f acc = {};
  for (int k0 = 0; k0 < 4096; k0 += 32) {
    v16h a_h = frag16(qhp + k0, half);
    v16h a_l = frag16(qlp + k0, half);
    v16h b_h = frag16(khp + k0, half);
    v16h b_l = frag16(klp + k0, half);
    acc = wmma3(acc, a_h, a_l, b_h, b_l);
  }
  __shared__ float S[32][33];
  float ts = temperature[h];
#pragma unroll
  for (int r = 0; r < 8; ++r) S[mt * 16 + half * 8 + r][nt * 16 + idx] = acc[r] * ts;
  __syncthreads();
  if (t < 32) {
    float mx = -3.4e38f;
    for (int j = 0; j < 32; ++j) mx = fmaxf(mx, S[t][j]);
    float e[32];
    float sum = 0.0f;
#pragma unroll
    for (int j = 0; j < 32; ++j) {
      e[j] = __expf(S[t][j] - mx);
      sum += e[j];
    }
    float inv = 1.0f / sum;
#pragma unroll
    for (int j = 0; j < 32; ++j) {
      float v = e[j] * inv;
      _Float16 hv = (_Float16)v;
      size_t o = (size_t)bh * 1024 + t * 32 + j;
      ath[o] = hv;
      atl[o] = (_Float16)(v - (float)hv);
    }
  }
}

// ---------------- out = attn @ v  (M=32, K=32, N=4096), all direct-global ----------------
// vT: [b][4096][256] (channel-contiguous), out written transposed [b][4096][256].
__global__ void __launch_bounds__(128) attnv_kernel(
    const _Float16* __restrict__ ath, const _Float16* __restrict__ atl,
    const _Float16* __restrict__ vhT, const _Float16* __restrict__ vlT,
    _Float16* __restrict__ ohT, _Float16* __restrict__ olT) {
  int n0 = blockIdx.x * 64;
  int bh = blockIdx.y;
  int b = bh >> 3, h = bh & 7;
  int t = threadIdx.x, wave = t >> 5, lane = t & 31, half = lane >> 4, idx = lane & 15;
  int mw = wave >> 1, nw = wave & 1;
  v16h a_h = frag16(ath + (size_t)bh * 1024 + (mw * 16 + idx) * 32, half);
  v16h a_l = frag16(atl + (size_t)bh * 1024 + (mw * 16 + idx) * 32, half);
#pragma unroll
  for (int s = 0; s < 2; ++s) {
    int n = n0 + nw * 32 + s * 16 + idx;
    size_t brow = ((size_t)b * 4096 + n) * 256 + h * 32;
    v16h b_h = frag16(vhT + brow, half);
    v16h b_l = frag16(vlT + brow, half);
    v8f acc = {};
    acc = wmma3(acc, a_h, a_l, b_h, b_l);
    int mb = mw * 16 + half * 8;
    v8h hv, lv;
#pragma unroll
    for (int r = 0; r < 8; ++r) {
      float v = acc[r];
      _Float16 hvv = (_Float16)v;
      hv[r] = hvv;
      lv[r] = (_Float16)(v - (float)hvv);
    }
    size_t o = ((size_t)b * 4096 + n) * 256 + h * 32 + mb;
    *(v8h*)(ohT + o) = hv;
    *(v8h*)(olT + o) = lv;
  }
}

// ---------------- host launcher ----------------
extern "C" void kernel_launch(void* const* d_in, const int* in_sizes, int n_in, void* d_out,
                              int out_size, void* d_ws, size_t ws_size, hipStream_t stream) {
  (void)in_sizes; (void)n_in; (void)out_size; (void)ws_size;
  const float* x = (const float*)d_in[0];
  const float* w_qkv = (const float*)d_in[1];
  const float* w_dw = (const float*)d_in[2];
  const float* temp = (const float*)d_in[3];
  const float* w_proj = (const float*)d_in[4];
  const float* w_mlp1 = (const float*)d_in[5];
  const float* b_mlp1 = (const float*)d_in[6];
  const float* w_mlp2 = (const float*)d_in[7];
  const float* b_mlp2 = (const float*)d_in[8];
  float* out = (float*)d_out;

  unsigned char* ws = (unsigned char*)d_ws;
  size_t off = 0;
  auto alloc = [&](size_t bytes) {
    size_t o = off;
    off = (off + bytes + 255) & ~(size_t)255;
    return o;
  };
  const size_t ACT = (size_t)8 * 256 * 4096;

  size_t o_wqkvh = alloc(768 * 256 * 2), o_wqkvl = alloc(768 * 256 * 2);
  size_t o_wprh = alloc(256 * 256 * 2), o_wprl = alloc(256 * 256 * 2);
  size_t o_w1h = alloc(320 * 256 * 2), o_w1l = alloc(320 * 256 * 2);
  size_t o_w2h = alloc(256 * 320 * 2), o_w2l = alloc(256 * 320 * 2);
  size_t o_ath = alloc(64 * 1024 * 2), o_atl = alloc(64 * 1024 * 2);
  size_t o_xh = alloc(ACT * 2), o_xl = alloc(ACT * 2);
  size_t o_A = alloc((size_t)8 * 768 * 4096 * 4);  // qkv_pre; later outT/x1/x1T
  size_t o_B = alloc((size_t)8 * 768 * 4096 * 4);  // qkv; later yT
  size_t o_C = alloc(6 * ACT * 2);                 // q,k (hi/lo) + vT (hi/lo)

  _Float16* wqkvh = (_Float16*)(ws + o_wqkvh);
  _Float16* wqkvl = (_Float16*)(ws + o_wqkvl);
  _Float16* wprh = (_Float16*)(ws + o_wprh);
  _Float16* wprl = (_Float16*)(ws + o_wprl);
  _Float16* w1h = (_Float16*)(ws + o_w1h);
  _Float16* w1l = (_Float16*)(ws + o_w1l);
  _Float16* w2h = (_Float16*)(ws + o_w2h);
  _Float16* w2l = (_Float16*)(ws + o_w2l);
  _Float16* ath = (_Float16*)(ws + o_ath);
  _Float16* atl = (_Float16*)(ws + o_atl);
  _Float16* xTh = (_Float16*)(ws + o_xh);   // [b][4096][256]
  _Float16* xTl = (_Float16*)(ws + o_xl);
  float* qkv_pre = (float*)(ws + o_A);
  float* qkvd = (float*)(ws + o_B);
  _Float16* qh = (_Float16*)(ws + o_C);
  _Float16* ql = qh + ACT;
  _Float16* kh = ql + ACT;
  _Float16* kl = kh + ACT;
  _Float16* vTh = kl + ACT;                 // [b][4096][256]
  _Float16* vTl = vTh + ACT;
  // region A reuse (qkv_pre dead after dwconv)
  _Float16* outTh = (_Float16*)(ws + o_A);  // [b][4096][256]
  _Float16* outTl = (_Float16*)(ws + o_A + ACT * 2);
  float* x1 = (float*)(ws + o_A + ACT * 4);        // [b][256][4096] f32
  _Float16* x1Th = (_Float16*)(ws + o_A + ACT * 8);   // [b][4096][256]
  _Float16* x1Tl = (_Float16*)(ws + o_A + ACT * 10);
  // region B reuse (qkv dead after norm/transpose)
  _Float16* yTh = (_Float16*)(ws + o_B);    // [b][4096][320]
  _Float16* yTl = (_Float16*)(ws + o_B + (size_t)8 * 4096 * 320 * 2);

  // P0: weight splits (+ zero-padding 307 -> 320) and x transpose-split
  split_kernel<<<768, 256, 0, stream>>>(w_qkv, wqkvh, wqkvl, (size_t)768 * 256);
  split_kernel<<<256, 256, 0, stream>>>(w_proj, wprh, wprl, (size_t)256 * 256);
  split_pad_rows<<<320, 256, 0, stream>>>(w_mlp1, w1h, w1l, 307, 256, 320);
  split_pad_cols<<<320, 256, 0, stream>>>(w_mlp2, w2h, w2l, 256, 307, 320);
  transpose_split<<<dim3(64, 4, 8), 256, 0, stream>>>(x, (size_t)256 * 4096, xTh, xTl);

  // P1: qkv = W_qkv @ x
  gemm_dg<false, false, false, true, false, 256><<<dim3(64, 12, 8), 256, 0, stream>>>(
      wqkvh, wqkvl, xTh, xTl, nullptr, nullptr, qkv_pre, nullptr, nullptr, 768, 768, 256);
  // P2: depthwise 3x3 (async plane staging)
  dwconv_kernel<<<8 * 768, 256, 0, stream>>>(qkv_pre, w_dw, qkvd);
  // P3: l2norm+split q,k ; transpose-split v
  norm_split_kernel<<<2048, 256, 0, stream>>>(qkvd, qh, ql, 0);
  norm_split_kernel<<<2048, 256, 0, stream>>>(qkvd, kh, kl, 1);
  transpose_split<<<dim3(64, 4, 8), 256, 0, stream>>>(qkvd + (size_t)512 * 4096,
                                                      (size_t)768 * 4096, vTh, vTl);
  // P4: attn = softmax(temperature * q k^T)
  attn_kernel<<<64, 128, 0, stream>>>(qh, ql, kh, kl, temp, ath, atl);
  // P5: outT = (attn @ v)^T
  attnv_kernel<<<dim3(64, 64), 128, 0, stream>>>(ath, atl, vTh, vTl, outTh, outTl);
  // P6: x1 = x + W_proj @ out   (f32 + transposed split)
  gemm_dg<false, false, true, true, true, 256><<<dim3(64, 4, 8), 256, 0, stream>>>(
      wprh, wprl, outTh, outTl, nullptr, x, x1, x1Th, x1Tl, 256, 256, 256);
  // P7: y = gelu(W_mlp1 @ x1 + b1)  (320 padded rows, transposed split only)
  gemm_dg<true, true, false, false, true, 256><<<dim3(64, 5, 8), 256, 0, stream>>>(
      w1h, w1l, x1Th, x1Tl, b_mlp1, nullptr, nullptr, yTh, yTl, 320, 307, 320);
  // P8: out = x1 + W_mlp2 @ y + b2  (K padded to 320)
  gemm_dg<true, false, true, true, false, 320><<<dim3(64, 4, 8), 256, 0, stream>>>(
      w2h, w2l, yTh, yTl, b_mlp2, x1, out, nullptr, nullptr, 256, 256, 320);
}
